// ATDTransformerLayer_9689446219854
// MI455X (gfx1250) — matrix operand order repaired and checked by reading, not compiled, not run on hardware
//
#include <hip/hip_runtime.h>
#include <math.h>

// ---------------- problem constants ----------------
#define BATCH   4
#define HH      128
#define WW      128
#define NTOKS   16384          // tokens per batch (HH*WW)
#define CDIM    192
#define NHEADS  6
#define HDIM    32
#define WSZ     16
#define WTOK    256            // tokens per window
#define NWIN    64             // windows per batch
#define NDICT   64
#define RDIM    10
#define DTD     64
#define MLPH    384
#define CH      448            // MLPH + DTD
#define QKV3    576            // 3*CDIM

typedef __attribute__((ext_vector_type(8)))  float   v8f;
typedef __attribute__((ext_vector_type(8)))  __bf16  v8bf;
typedef __attribute__((ext_vector_type(16))) __bf16  v16bf;

// ---------------- helpers ----------------
__device__ inline v8f wmma_bf16(v16bf a, v16bf b, v8f c) {
  return __builtin_amdgcn_wmma_f32_16x16x32_bf16(false, a, false, b, (short)0, c, false, false);
}

__device__ inline float gelu_f(float x) {
  return 0.5f * x * (1.0f + erff(x * 0.70710678118654752f));
}

// A-operand: 16x32 bf16 tile, row-major with leading dim ld.
// Per ISA layout, lane(m,hi) needs K = hi*8+0..7 (v0..3) and 16+hi*8+0..7 (v4..7):
// two contiguous 16-byte loads per lane.
__device__ inline v16bf load_a16v(const __bf16* base, int ld, int lane) {
  int m = lane & 15;
  int hi = (lane >> 4) & 1;
  const __bf16* p = base + m * ld + hi * 8;
  v8bf lo = *(const v8bf*)p;
  v8bf hh = *(const v8bf*)(p + 16);
  return __builtin_shufflevector(lo, hh, 0, 1, 2, 3, 4, 5, 6, 7,
                                 8, 9, 10, 11, 12, 13, 14, 15);
}

// B-operand from K-major storage: B[k][n] = base[n*ld + k]; lane(n,kh) needs
// K = kh*8+0..7 and 16+kh*8+0..7: two contiguous 16-byte loads.
__device__ inline v16bf load_bTv(const __bf16* base, int ld, int lane) {
  int n  = lane & 15;
  int kh = (lane >> 4) & 1;
  const __bf16* p = base + n * ld + kh * 8;
  v8bf lo = *(const v8bf*)p;
  v8bf hh = *(const v8bf*)(p + 16);
  return __builtin_shufflevector(lo, hh, 0, 1, 2, 3, 4, 5, 6, 7,
                                 8, 9, 10, 11, 12, 13, 14, 15);
}

__device__ inline float half16_max(float x) {
#pragma unroll
  for (int off = 1; off < 16; off <<= 1) x = fmaxf(x, __shfl_xor(x, off, 32));
  return x;
}
__device__ inline float half16_sum(float x) {
#pragma unroll
  for (int off = 1; off < 16; off <<= 1) x += __shfl_xor(x, off, 32);
  return x;
}

// ---------------- elementwise utility kernels ----------------
__global__ void k_f2bf(const float* __restrict__ s, __bf16* __restrict__ d, long n) {
  long i = (long)blockIdx.x * blockDim.x + threadIdx.x;
  if (i < n) d[i] = (__bf16)s[i];
}
__global__ void k_copy_f32(const float* __restrict__ s, float* __restrict__ d, long n) {
  long i = (long)blockIdx.x * blockDim.x + threadIdx.x;
  if (i < n) d[i] = s[i];
}
__global__ void k_zero_i32(int* __restrict__ d, long n) {
  long i = (long)blockIdx.x * blockDim.x + threadIdx.x;
  if (i < n) d[i] = 0;
}

// ---------------- LayerNorm (block of 192 threads per token) ----------------
__global__ void k_layernorm(const float* __restrict__ x, const float* __restrict__ g,
                            const float* __restrict__ b, __bf16* __restrict__ out) {
  __shared__ float part[6];
  __shared__ float stat;
  long row = blockIdx.x;
  int tid = threadIdx.x, lane = tid & 31, wid = tid >> 5;
  float v = x[row * CDIM + tid];
  float s = v;
#pragma unroll
  for (int off = 16; off; off >>= 1) s += __shfl_xor(s, off, 32);
  if (lane == 0) part[wid] = s;
  __syncthreads();
  if (tid == 0) { float m = 0; for (int i = 0; i < 6; ++i) m += part[i]; stat = m * (1.0f / CDIM); }
  __syncthreads();
  float mu = stat;
  float d  = v - mu;
  float s2 = d * d;
#pragma unroll
  for (int off = 16; off; off >>= 1) s2 += __shfl_xor(s2, off, 32);
  __syncthreads();
  if (lane == 0) part[wid] = s2;
  __syncthreads();
  if (tid == 0) { float m = 0; for (int i = 0; i < 6; ++i) m += part[i]; stat = m * (1.0f / CDIM); }
  __syncthreads();
  float r = rsqrtf(stat + 1e-5f);
  out[row * CDIM + tid] = (__bf16)(d * r * g[tid] + b[tid]);
}

// ---------------- generic WMMA GEMM: C = A(MxK) * W(KxN) [+bias][gelu] ----------------
// 256 threads, 64x64 tile, 8 waves each computing 16x32 of C with K-steps of 32.
// W tile stored transposed in LDS so all WMMA operands use contiguous b128 LDS loads.
__global__ void k_gemm_wmma(const __bf16* __restrict__ A, long bsA,
                            const __bf16* __restrict__ W, long bsW,
                            const float* __restrict__ bias,
                            float* outF, long bsOF, int ldoF, int accF,
                            __bf16* outB, long bsOB, int ldoB,
                            int M, int N, int K, int gelu) {
  __shared__ __bf16 Al[64 * 32];       // [m][k]
  __shared__ __bf16 Wlt[64 * 32];      // [n][k]  (transposed)
  int bz = blockIdx.z;
  const __bf16* Ab = A + bsA * bz;
  const __bf16* Wb = W + bsW * bz;
  int bm = blockIdx.y * 64, bn = blockIdx.x * 64;
  int tid = threadIdx.x, lane = tid & 31, wid = tid >> 5;
  int mo = (wid & 3) * 16, no = (wid >> 2) * 32;
  int r = tid >> 2, s = (tid & 3) * 8;       // A staging: 64 rows x 32 k
  int rw = tid >> 3, sw = (tid & 7) * 8;     // W staging: 32 k x 64 n
  v8f c0 = {}, c1 = {};
  for (int k0 = 0; k0 < K; k0 += 32) {
    *(uint4*)&Al[r * 32 + s] = *(const uint4*)&Ab[(long)(bm + r) * K + k0 + s];
    { v8bf wv = *(const v8bf*)&Wb[(long)(k0 + rw) * N + bn + sw];
#pragma unroll
      for (int e = 0; e < 8; ++e) Wlt[(sw + e) * 32 + rw] = wv[e]; }
    if (k0 + 32 < K) {
      __builtin_prefetch(&Ab[(long)(bm + r) * K + k0 + 32 + s], 0, 1);
      __builtin_prefetch(&Wb[(long)(k0 + 32 + rw) * N + bn + sw], 0, 1);
    }
    __syncthreads();
    v16bf a  = load_a16v(&Al[mo * 32], 32, lane);
    v16bf b0 = load_bTv(&Wlt[no * 32], 32, lane);
    v16bf b1 = load_bTv(&Wlt[(no + 16) * 32], 32, lane);
    c0 = wmma_bf16(a, b0, c0);
    c1 = wmma_bf16(a, b1, c1);
    __syncthreads();
  }
  int n = lane & 15, half = (lane >> 4) & 1;
#pragma unroll
  for (int v = 0; v < 8; ++v) {
    int row  = bm + mo + v + half * 8;
    int col0 = bn + no + n, col1 = col0 + 16;
    float v0 = c0[v], v1 = c1[v];
    if (bias) { v0 += bias[col0]; v1 += bias[col1]; }
    if (gelu) { v0 = gelu_f(v0); v1 = gelu_f(v1); }
    if (outF) {
      float* p = outF + bsOF * bz + (long)row * ldoF;
      if (accF) { p[col0] += v0; p[col1] += v1; } else { p[col0] = v0; p[col1] = v1; }
    }
    if (outB) {
      __bf16* p = outB + bsOB * bz + (long)row * ldoB;
      p[col0] = (__bf16)v0; p[col1] = (__bf16)v1;
    }
  }
}

// ---------------- small dense GEMM (N=10 paths), fp32 VALU ----------------
__global__ void k_smallgemm(const __bf16* __restrict__ A, const float* __restrict__ W,
                            const float* __restrict__ bias, float* __restrict__ out,
                            long total, int N, int K) {
  long idx = (long)blockIdx.x * blockDim.x + threadIdx.x;
  if (idx >= total) return;
  int j = (int)(idx % N);
  long row = idx / N;
  float acc = bias[j];
  for (int k = 0; k < K; ++k) acc += (float)A[row * K + k] * W[k * N + j];
  out[idx] = acc;
}

__global__ void k_knorm(const float* __restrict__ kd, float* __restrict__ kn) {
  int r = blockIdx.x * blockDim.x + threadIdx.x;
  if (r >= BATCH * NDICT) return;
  const float* p = kd + (long)r * RDIM;
  float s = 0;
  for (int i = 0; i < RDIM; ++i) s += p[i] * p[i];
  float inv = 1.0f / fmaxf(sqrtf(s), 1e-12f);
  for (int i = 0; i < RDIM; ++i) kn[(long)r * RDIM + i] = p[i] * inv;
}

// ---------------- ATD cross-attention: one wave per token ----------------
__global__ void k_sim(const float* __restrict__ qf, const float* __restrict__ kn,
                      const float* __restrict__ scl, __bf16* __restrict__ simb,
                      int* __restrict__ tkid) {
  long widg = ((long)blockIdx.x * blockDim.x + threadIdx.x) >> 5;
  int lane = threadIdx.x & 31;
  if (widg >= (long)BATCH * NTOKS) return;
  int b = (int)(widg / NTOKS);
  const float* q = qf + widg * RDIM;
  float qv[RDIM]; float nrm = 0;
#pragma unroll
  for (int i = 0; i < RDIM; ++i) { qv[i] = q[i]; nrm += qv[i] * qv[i]; }
  nrm = 1.0f / fmaxf(sqrtf(nrm), 1e-12f);
#pragma unroll
  for (int i = 0; i < RDIM; ++i) qv[i] *= nrm;
  float sc = fminf(fmaxf(scl[0], 0.0f), 3.0f);
  sc = 1.0f + sc * 4.1588830833596715f;   // log(64)
  const float* k0 = kn + (long)b * NDICT * RDIM + lane * RDIM;
  const float* k1 = k0 + 32 * RDIM;
  float s0 = 0, s1 = 0;
#pragma unroll
  for (int i = 0; i < RDIM; ++i) { s0 += qv[i] * k0[i]; s1 += qv[i] * k1[i]; }
  // argmax (stable: lowest index on ties)
  float bv = s0; int bi = lane;
  if (s1 > bv) { bv = s1; bi = lane + 32; }
#pragma unroll
  for (int off = 16; off; off >>= 1) {
    float ov = __shfl_xor(bv, off, 32);
    int   oi = __shfl_xor(bi, off, 32);
    if (ov > bv || (ov == bv && oi < bi)) { bv = ov; bi = oi; }
  }
  if (lane == 0) tkid[widg] = bi;
  float l0 = s0 * sc, l1 = s1 * sc;
  float mx = fmaxf(l0, l1);
#pragma unroll
  for (int off = 16; off; off >>= 1) mx = fmaxf(mx, __shfl_xor(mx, off, 32));
  float e0 = __expf(l0 - mx), e1 = __expf(l1 - mx);
  float sm = e0 + e1;
#pragma unroll
  for (int off = 16; off; off >>= 1) sm += __shfl_xor(sm, off, 32);
  float inv = 1.0f / sm;
  simb[widg * NDICT + lane]      = (__bf16)(e0 * inv);
  simb[widg * NDICT + lane + 32] = (__bf16)(e1 * inv);
}

// ---------------- stable counting sort (wave32 ballots, no atomics) ----------------
__device__ inline unsigned match_key(int key) {
  unsigned m = 0xffffffffu;
#pragma unroll
  for (int bit = 0; bit < 6; ++bit) {
    unsigned bal = (unsigned)__ballot((key >> bit) & 1);
    m &= ((key >> bit) & 1) ? bal : ~bal;
  }
  return m;
}

__global__ void k_sort_count(const int* __restrict__ tkid, int* __restrict__ wavehist) {
  long gt = (long)blockIdx.x * blockDim.x + threadIdx.x;
  if (gt >= (long)BATCH * NTOKS) return;
  int b = (int)(gt / NTOKS), t = (int)(gt % NTOKS);
  int wv = t >> 5, lane = threadIdx.x & 31;
  int key = tkid[gt];
  unsigned mask = match_key(key);
  unsigned lt = (1u << lane) - 1u;
  if ((mask & lt) == 0) wavehist[((long)b * 512 + wv) * 64 + key] = __popc(mask);
}

__global__ void k_sort_scanwaves(const int* __restrict__ wavehist, int* __restrict__ waveoff,
                                 int* __restrict__ keycnt) {
  int tid = blockIdx.x * blockDim.x + threadIdx.x;
  if (tid >= BATCH * 64) return;
  int b = tid / 64, key = tid % 64;
  int run = 0;
  for (int wv = 0; wv < 512; ++wv) {
    long idx = ((long)b * 512 + wv) * 64 + key;
    waveoff[idx] = run;
    run += wavehist[idx];
  }
  keycnt[b * 64 + key] = run;
}

__global__ void k_sort_scankeys(const int* __restrict__ keycnt, int* __restrict__ keybase) {
  int b = threadIdx.x;
  if (b >= BATCH) return;
  int base = 0;
  for (int key = 0; key < 64; ++key) { keybase[b * 64 + key] = base; base += keycnt[b * 64 + key]; }
}

__global__ void k_sort_place(const int* __restrict__ tkid, const int* __restrict__ keybase,
                             const int* __restrict__ waveoff, int* __restrict__ sidx,
                             int* __restrict__ invp) {
  long gt = (long)blockIdx.x * blockDim.x + threadIdx.x;
  if (gt >= (long)BATCH * NTOKS) return;
  int b = (int)(gt / NTOKS), t = (int)(gt % NTOKS);
  int wv = t >> 5, lane = threadIdx.x & 31;
  int key = tkid[gt];
  unsigned mask = match_key(key);
  unsigned lt = (1u << lane) - 1u;
  int rank = __popc(mask & lt);
  int pos = keybase[b * 64 + key] + waveoff[((long)b * 512 + wv) * 64 + key] + rank;
  sidx[(long)b * NTOKS + pos] = t;
  invp[(long)b * NTOKS + t]  = pos;
}

// ---------------- AC_MSA: 128x128 attention per (b, group, head) ----------------
__global__ void k_ac_attn(const __bf16* __restrict__ qkv, const int* __restrict__ sidx,
                          __bf16* __restrict__ ysort) {
  __shared__ __bf16 Kl[128 * 32];      // [key][hd]
  __shared__ __bf16 Vt[32 * 128];      // [hd][key]  (transposed)
  __shared__ __bf16 Ps[8 * 16 * 128];
  int bid = blockIdx.x;
  int h = bid % NHEADS, g = (bid / NHEADS) % 128, b = bid / (NHEADS * 128);
  int tid = threadIdx.x, lane = tid & 31, wid = tid >> 5;
  const long qbase = (long)b * NTOKS * QKV3;
  { int r = tid >> 1, hf = (tid & 1) * 16;
    int tok = sidx[(long)b * NTOKS + g * 128 + r];
    const __bf16* kp = qkv + qbase + (long)tok * QKV3 + CDIM + h * HDIM + hf;
    const __bf16* vp = qkv + qbase + (long)tok * QKV3 + 2 * CDIM + h * HDIM + hf;
    *(uint4*)&Kl[r * 32 + hf]     = *(const uint4*)kp;
    *(uint4*)&Kl[r * 32 + hf + 8] = *(const uint4*)(kp + 8);
    v8bf v0 = *(const v8bf*)vp;
    v8bf v1 = *(const v8bf*)(vp + 8);
#pragma unroll
    for (int e = 0; e < 8; ++e) {
      Vt[(hf + e) * 128 + r]     = v0[e];
      Vt[(hf + 8 + e) * 128 + r] = v1[e];
    } }
  __syncthreads();
  int rowbase = wid * 16;
  v16bf qa;
  { int m = lane & 15, hi = (lane >> 4) & 1;
    int tok = sidx[(long)b * NTOKS + g * 128 + rowbase + m];
    const __bf16* qp = qkv + qbase + (long)tok * QKV3 + h * HDIM + hi * 8;
    v8bf lo = *(const v8bf*)qp;
    v8bf hh = *(const v8bf*)(qp + 16);
    qa = __builtin_shufflevector(lo, hh, 0, 1, 2, 3, 4, 5, 6, 7,
                                 8, 9, 10, 11, 12, 13, 14, 15); }
  v8f S[8];
#pragma unroll
  for (int kt = 0; kt < 8; ++kt) {
    v16bf bt = load_bTv(&Kl[kt * 16 * 32], 32, lane);
    v8f z = {};
    S[kt] = wmma_bf16(qa, bt, z);
  }
  const float scale = 0.17677669529663687f; // 32^-0.5
  float rs[8];
#pragma unroll
  for (int v = 0; v < 8; ++v) {
    float mx = -1e30f;
#pragma unroll
    for (int kt = 0; kt < 8; ++kt) { S[kt][v] *= scale; mx = fmaxf(mx, S[kt][v]); }
    mx = half16_max(mx);
    float sum = 0;
#pragma unroll
    for (int kt = 0; kt < 8; ++kt) { S[kt][v] = __expf(S[kt][v] - mx); sum += S[kt][v]; }
    rs[v] = half16_sum(sum);
  }
  __bf16* scr = &Ps[wid * 16 * 128];
  int n = lane & 15, half = (lane >> 4) & 1;
#pragma unroll
  for (int kt = 0; kt < 8; ++kt)
#pragma unroll
    for (int v = 0; v < 8; ++v) {
      int r = v + half * 8;
      scr[r * 128 + kt * 16 + n] = (__bf16)S[kt][v];
    }
  v8f y0 = {}, y1 = {};
#pragma unroll
  for (int c = 0; c < 4; ++c) {
    v16bf pa = load_a16v(scr + c * 32, 128, lane);
    v16bf b0 = load_bTv(&Vt[c * 32], 128, lane);               // hd cols 0..15
    v16bf b1 = load_bTv(&Vt[16 * 128 + c * 32], 128, lane);    // hd cols 16..31
    y0 = wmma_bf16(pa, b0, y0);
    y1 = wmma_bf16(pa, b1, y1);
  }
#pragma unroll
  for (int v = 0; v < 8; ++v) {
    float inv = 1.0f / rs[v];
    int r = rowbase + v + half * 8;
    long pos = (long)b * NTOKS + g * 128 + r;
    ysort[pos * CDIM + h * HDIM + n]      = (__bf16)(y0[v] * inv);
    ysort[pos * CDIM + h * HDIM + 16 + n] = (__bf16)(y1[v] * inv);
  }
}

__global__ void k_gather_rows(const __bf16* __restrict__ ysort, const int* __restrict__ invp,
                              __bf16* __restrict__ yuns) {
  long idx = (long)blockIdx.x * blockDim.x + threadIdx.x;
  long total = (long)BATCH * NTOKS * CDIM;
  if (idx >= total) return;
  int c = (int)(idx % CDIM);
  long bi = idx / CDIM;
  int i = (int)(bi % NTOKS);
  int b = (int)(bi / NTOKS);
  int p = invp[(long)b * NTOKS + i];
  yuns[idx] = ysort[((long)b * NTOKS + p) * CDIM + c];
}

// ---------------- shifted-window attention (flash-style, online softmax) ----------------
__global__ void k_win_attn(const __bf16* __restrict__ qkv, const float* __restrict__ mask,
                           const int* __restrict__ rpi, const float* __restrict__ rpb,
                           __bf16* __restrict__ yout) {
  __shared__ __bf16 Kl[256 * 32];      // [key][hd]
  __shared__ __bf16 Vt[32 * 256];      // [hd][key]  (transposed)
  __shared__ __bf16 Ps[16 * 16 * 32];
  int bid = blockIdx.x;
  int h = bid % NHEADS, wi = (bid / NHEADS) % NWIN, b = bid / (NHEADS * NWIN);
  int wr = wi >> 3, wc = wi & 7;
  int tid = threadIdx.x, lane = tid & 31, wid = tid >> 5;  // 512 threads, 16 waves
  const long qbase = (long)b * NTOKS * QKV3;
#define ORIG_TOK(t) ({ int _i = (t) >> 4, _j = (t) & 15; \
                       int _r = (wr * 16 + _i + 8) & 127; \
                       int _c = (wc * 16 + _j + 8) & 127; \
                       _r * 128 + _c; })
  { int t = tid >> 1, hf = (tid & 1) * 16;
    int tok = ORIG_TOK(t);
    const __bf16* kp = qkv + qbase + (long)tok * QKV3 + CDIM + h * HDIM + hf;
    const __bf16* vp = qkv + qbase + (long)tok * QKV3 + 2 * CDIM + h * HDIM + hf;
    *(uint4*)&Kl[t * 32 + hf]     = *(const uint4*)kp;
    *(uint4*)&Kl[t * 32 + hf + 8] = *(const uint4*)(kp + 8);
    v8bf v0 = *(const v8bf*)vp;
    v8bf v1 = *(const v8bf*)(vp + 8);
#pragma unroll
    for (int e = 0; e < 8; ++e) {
      Vt[(hf + e) * 256 + t]     = v0[e];
      Vt[(hf + 8 + e) * 256 + t] = v1[e];
    } }
  __syncthreads();
  int rowbase = wid * 16;
  v16bf qa;
  { int m = lane & 15, hi = (lane >> 4) & 1;
    int tok = ORIG_TOK(rowbase + m);
    const __bf16* qp = qkv + qbase + (long)tok * QKV3 + h * HDIM + hi * 8;
    v8bf lo = *(const v8bf*)qp;
    v8bf hh = *(const v8bf*)(qp + 16);
    qa = __builtin_shufflevector(lo, hh, 0, 1, 2, 3, 4, 5, 6, 7,
                                 8, 9, 10, 11, 12, 13, 14, 15); }
  float rm[8], rs[8];
#pragma unroll
  for (int v = 0; v < 8; ++v) { rm[v] = -1e30f; rs[v] = 0.0f; }
  v8f y0 = {}, y1 = {};
  __bf16* scr = &Ps[wid * 512];
  int n = lane & 15, half = (lane >> 4) & 1;
  const float scale = 0.17677669529663687f;
  for (int ck = 0; ck < 8; ++ck) {
    int k0 = ck * 32;
    v16bf bt0 = load_bTv(&Kl[k0 * 32], 32, lane);
    v16bf bt1 = load_bTv(&Kl[(k0 + 16) * 32], 32, lane);
    v8f z = {};
    v8f s0 = wmma_bf16(qa, bt0, z);
    v8f s1 = wmma_bf16(qa, bt1, z);
#pragma unroll
    for (int v = 0; v < 8; ++v) {
      int qrow = rowbase + v + half * 8;
      int kc0 = k0 + n, kc1 = k0 + 16 + n;
      float bias0 = rpb[rpi[qrow * WTOK + kc0] * NHEADS + h] +
                    mask[((long)wi * WTOK + qrow) * WTOK + kc0];
      float bias1 = rpb[rpi[qrow * WTOK + kc1] * NHEADS + h] +
                    mask[((long)wi * WTOK + qrow) * WTOK + kc1];
      s0[v] = s0[v] * scale + bias0;
      s1[v] = s1[v] * scale + bias1;
    }
#pragma unroll
    for (int v = 0; v < 8; ++v) {
      float cm = half16_max(fmaxf(s0[v], s1[v]));
      float nm = fmaxf(rm[v], cm);
      float f  = __expf(rm[v] - nm);
      s0[v] = __expf(s0[v] - nm);
      s1[v] = __expf(s1[v] - nm);
      float cs = half16_sum(s0[v] + s1[v]);
      rs[v] = rs[v] * f + cs;
      rm[v] = nm;
      y0[v] *= f; y1[v] *= f;
      int r = v + half * 8;
      scr[r * 32 + n]      = (__bf16)s0[v];
      scr[r * 32 + 16 + n] = (__bf16)s1[v];
    }
    v16bf pa  = load_a16v(scr, 32, lane);
    v16bf bv0 = load_bTv(&Vt[k0], 256, lane);                 // hd cols 0..15
    v16bf bv1 = load_bTv(&Vt[16 * 256 + k0], 256, lane);      // hd cols 16..31
    y0 = wmma_bf16(pa, bv0, y0);
    y1 = wmma_bf16(pa, bv1, y1);
  }
#pragma unroll
  for (int v = 0; v < 8; ++v) {
    float inv = 1.0f / rs[v];
    int t = rowbase + v + half * 8;
    long tok = ORIG_TOK(t);
    long o = ((long)b * NTOKS + tok) * CDIM + h * HDIM;
    yout[o + n]      = (__bf16)(y0[v] * inv);
    yout[o + 16 + n] = (__bf16)(y1[v] * inv);
  }
#undef ORIG_TOK
}

// ---------------- FFN extras ----------------
__global__ void k_xtd_gather(const float* __restrict__ tdf, const int* __restrict__ tkid,
                             __bf16* __restrict__ xc) {
  long idx = (long)blockIdx.x * blockDim.x + threadIdx.x;
  long total = (long)BATCH * NTOKS * DTD;
  if (idx >= total) return;
  int c = (int)(idx % DTD);
  long bi = idx / DTD;
  int i = (int)(bi % NTOKS);
  int b = (int)(bi / NTOKS);
  int tk = tkid[(long)b * NTOKS + i];
  xc[((long)b * NTOKS + i) * CH + MLPH + c] = (__bf16)tdf[((long)b * NDICT + tk) * DTD + c];
}

__global__ void k_dwconv(const __bf16* __restrict__ xc, const float* __restrict__ w,
                         const float* __restrict__ bias, __bf16* __restrict__ out) {
  long idx = (long)blockIdx.x * blockDim.x + threadIdx.x;
  long total = (long)BATCH * NTOKS * CH;
  if (idx >= total) return;
  int ch = (int)(idx % CH);
  long pix = idx / CH;
  int x2 = (int)(pix % WW);
  long tmp = pix / WW;
  int y2 = (int)(tmp % HH);
  int b  = (int)(tmp / HH);
  float acc = 0;
#pragma unroll
  for (int ky = 0; ky < 5; ++ky) {
    int yy = y2 + ky - 2;
    if (yy < 0 || yy >= HH) continue;
#pragma unroll
    for (int kx = 0; kx < 5; ++kx) {
      int xx = x2 + kx - 2;
      if (xx < 0 || xx >= WW) continue;
      acc += (float)xc[(((long)b * HH + yy) * WW + xx) * CH + ch] * w[(ky * 5 + kx) * CH + ch];
    }
  }
  float cv = gelu_f(acc + bias[ch]);
  out[idx] = (__bf16)((float)xc[idx] + cv);
}

// ---------------- host launch ----------------
extern "C" void kernel_launch(void* const* d_in, const int* in_sizes, int n_in,
                              void* d_out, int out_size, void* d_ws, size_t ws_size,
                              hipStream_t stream) {
  const float* x         = (const float*)d_in[0];
  const float* td        = (const float*)d_in[1];
  const float* attn_mask = (const float*)d_in[2];
  const int*   rpi       = (const int*)d_in[3];
  const float* n1g = (const float*)d_in[6];
  const float* n1b = (const float*)d_in[7];
  const float* n2g = (const float*)d_in[8];
  const float* n2b = (const float*)d_in[9];
  const float* wqkv_w = (const float*)d_in[10];
  const float* wqkv_b = (const float*)d_in[11];
  const float* wq_w = (const float*)d_in[12];
  const float* wq_b = (const float*)d_in[13];
  const float* wk_w = (const float*)d_in[14];
  const float* wk_b = (const float*)d_in[15];
  const float* wv_w = (const float*)d_in[16];
  const float* wv_b = (const float*)d_in[17];
  const float* atd_scale = (const float*)d_in[18];
  const float* aca_w = (const float*)d_in[19];
  const float* aca_b = (const float*)d_in[20];
  const float* rpb   = (const float*)d_in[21];
  const float* winp_w = (const float*)d_in[22];
  const float* winp_b = (const float*)d_in[23];
  const float* fctd_w = (const float*)d_in[24];
  const float* fctd_b = (const float*)d_in[25];
  const float* fc1_w = (const float*)d_in[26];
  const float* fc1_b = (const float*)d_in[27];
  const float* dw_w  = (const float*)d_in[28];
  const float* dw_b  = (const float*)d_in[29];
  const float* fc2_w = (const float*)d_in[30];
  const float* fc2_b = (const float*)d_in[31];

  const long BN = (long)BATCH * NTOKS;   // 65536

  size_t off = 0;
  auto alloc = [&](size_t bytes) -> void* {
    void* p = (char*)d_ws + off;
    off += (bytes + 255) & ~(size_t)255;
    return p;
  };
  __bf16* xn_bf   = (__bf16*)alloc(BN * CDIM * 2);
  __bf16* xn2_bf  = (__bf16*)alloc(BN * CDIM * 2);
  __bf16* qkv_bf  = (__bf16*)alloc(BN * QKV3 * 2);
  __bf16* td_bf   = (__bf16*)alloc((size_t)BATCH * NDICT * CDIM * 2);
  __bf16* wqkv_bf = (__bf16*)alloc((size_t)CDIM * QKV3 * 2);
  __bf16* wv_bf   = (__bf16*)alloc((size_t)CDIM * CDIM * 2);
  __bf16* acaw_bf = (__bf16*)alloc((size_t)CDIM * CDIM * 2);
  __bf16* winw_bf = (__bf16*)alloc((size_t)CDIM * CDIM * 2);
  __bf16* fctd_bf = (__bf16*)alloc((size_t)CDIM * DTD * 2);
  __bf16* fc1_bf  = (__bf16*)alloc((size_t)CDIM * MLPH * 2);
  __bf16* fc2_bf  = (__bf16*)alloc((size_t)CH * CDIM * 2);
  float*  q_f     = (float*)alloc(BN * RDIM * 4);
  float*  kd_f    = (float*)alloc((size_t)BATCH * NDICT * RDIM * 4);
  float*  kn_f    = (float*)alloc((size_t)BATCH * NDICT * RDIM * 4);
  __bf16* vd_bf   = (__bf16*)alloc((size_t)BATCH * NDICT * CDIM * 2);
  float*  tdf_f   = (float*)alloc((size_t)BATCH * NDICT * DTD * 4);
  __bf16* sim_bf  = (__bf16*)alloc(BN * NDICT * 2);
  int*    tkid    = (int*)alloc(BN * 4);
  int*    wavehist= (int*)alloc((size_t)BATCH * 512 * 64 * 4);
  int*    waveoff = (int*)alloc((size_t)BATCH * 512 * 64 * 4);
  int*    keycnt  = (int*)alloc((size_t)BATCH * 64 * 4);
  int*    keybase = (int*)alloc((size_t)BATCH * 64 * 4);
  int*    sidx    = (int*)alloc(BN * 4);
  int*    invp    = (int*)alloc(BN * 4);
  __bf16* ysort   = (__bf16*)alloc(BN * CDIM * 2);
  __bf16* yuns    = (__bf16*)alloc(BN * CDIM * 2);
  __bf16* ywin    = (__bf16*)alloc(BN * CDIM * 2);
  float*  x_res   = (float*)alloc(BN * CDIM * 4);
  __bf16* xc_bf   = (__bf16*)alloc(BN * CH * 2);
  __bf16* s_bf    = (__bf16*)alloc(BN * CH * 2);

  auto cdiv = [](long a, long b) { return (int)((a + b - 1) / b); };
  float* outF = (float*)d_out;

  // weight / input conversions to bf16
  k_f2bf<<<cdiv((long)CDIM * QKV3, 256), 256, 0, stream>>>(wqkv_w, wqkv_bf, (long)CDIM * QKV3);
  k_f2bf<<<cdiv((long)CDIM * CDIM, 256), 256, 0, stream>>>(wv_w, wv_bf, (long)CDIM * CDIM);
  k_f2bf<<<cdiv((long)CDIM * CDIM, 256), 256, 0, stream>>>(aca_w, acaw_bf, (long)CDIM * CDIM);
  k_f2bf<<<cdiv((long)CDIM * CDIM, 256), 256, 0, stream>>>(winp_w, winw_bf, (long)CDIM * CDIM);
  k_f2bf<<<cdiv((long)CDIM * DTD, 256), 256, 0, stream>>>(fctd_w, fctd_bf, (long)CDIM * DTD);
  k_f2bf<<<cdiv((long)CDIM * MLPH, 256), 256, 0, stream>>>(fc1_w, fc1_bf, (long)CDIM * MLPH);
  k_f2bf<<<cdiv((long)CH * CDIM, 256), 256, 0, stream>>>(fc2_w, fc2_bf, (long)CH * CDIM);
  k_f2bf<<<cdiv((long)BATCH * NDICT * CDIM, 256), 256, 0, stream>>>(td, td_bf, (long)BATCH * NDICT * CDIM);

  // LN1 -> xn
  k_layernorm<<<BN, CDIM, 0, stream>>>(x, n1g, n1b, xn_bf);

  // qkv = xn @ wqkv + b  (bf16 out)
  k_gemm_wmma<<<dim3(QKV3 / 64, (int)(BN / 64), 1), 256, 0, stream>>>(
      xn_bf, 0, wqkv_bf, 0, wqkv_b, nullptr, 0, 0, 0, qkv_bf, 0, QKV3,
      (int)BN, QKV3, CDIM, 0);

  // ATD: q, k, v, td_f
  k_smallgemm<<<cdiv(BN * RDIM, 256), 256, 0, stream>>>(xn_bf, wq_w, wq_b, q_f, BN * RDIM, RDIM, CDIM);
  k_smallgemm<<<cdiv((long)BATCH * NDICT * RDIM, 256), 256, 0, stream>>>(
      td_bf, wk_w, wk_b, kd_f, (long)BATCH * NDICT * RDIM, RDIM, CDIM);
  k_knorm<<<1, 256, 0, stream>>>(kd_f, kn_f);
  k_gemm_wmma<<<dim3(CDIM / 64, 1, BATCH), 256, 0, stream>>>(
      td_bf, (long)NDICT * CDIM, wv_bf, 0, wv_b, nullptr, 0, 0, 0,
      vd_bf, (long)NDICT * CDIM, CDIM, NDICT, CDIM, CDIM, 0);
  k_gemm_wmma<<<dim3(1, 1, BATCH), 256, 0, stream>>>(
      td_bf, (long)NDICT * CDIM, fctd_bf, 0, fctd_b, tdf_f, (long)NDICT * DTD, DTD, 0,
      nullptr, 0, 0, NDICT, DTD, CDIM, 0);

  // sim + tk_id
  k_sim<<<cdiv(BN * 32, 256), 256, 0, stream>>>(q_f, kn_f, atd_scale, sim_bf, tkid);

  // x_res = shortcut
  k_copy_f32<<<cdiv(BN * CDIM, 256), 256, 0, stream>>>(x, x_res, BN * CDIM);

  // x_res += sim @ v   (batched, f32 accumulate)
  k_gemm_wmma<<<dim3(CDIM / 64, NTOKS / 64, BATCH), 256, 0, stream>>>(
      sim_bf, (long)NTOKS * NDICT, vd_bf, (long)NDICT * CDIM, nullptr,
      x_res, (long)NTOKS * CDIM, CDIM, 1, nullptr, 0, 0, NTOKS, CDIM, NDICT, 0);

  // window attention -> ywin ; x_res += ywin @ win_proj + b
  k_win_attn<<<BATCH * NWIN * NHEADS, 512, 0, stream>>>(qkv_bf, attn_mask, rpi, rpb, ywin);
  k_gemm_wmma<<<dim3(CDIM / 64, (int)(BN / 64), 1), 256, 0, stream>>>(
      ywin, 0, winw_bf, 0, winp_b, x_res, 0, CDIM, 1, nullptr, 0, 0,
      (int)BN, CDIM, CDIM, 0);

  // stable counting sort of tk_id
  k_zero_i32<<<cdiv((long)BATCH * 512 * 64, 256), 256, 0, stream>>>(wavehist, (long)BATCH * 512 * 64);
  k_sort_count<<<cdiv(BN, 256), 256, 0, stream>>>(tkid, wavehist);
  k_sort_scanwaves<<<1, 256, 0, stream>>>(wavehist, waveoff, keycnt);
  k_sort_scankeys<<<1, BATCH, 0, stream>>>(keycnt, keybase);
  k_sort_place<<<cdiv(BN, 256), 256, 0, stream>>>(tkid, keybase, waveoff, sidx, invp);

  // AC_MSA -> ysort -> unsort -> x_res += yuns @ aca_proj + b
  k_ac_attn<<<BATCH * (NTOKS / 128) * NHEADS, 256, 0, stream>>>(qkv_bf, sidx, ysort);
  k_gather_rows<<<cdiv(BN * CDIM, 256), 256, 0, stream>>>(ysort, invp, yuns);
  k_gemm_wmma<<<dim3(CDIM / 64, (int)(BN / 64), 1), 256, 0, stream>>>(
      yuns, 0, acaw_bf, 0, aca_b, x_res, 0, CDIM, 1, nullptr, 0, 0,
      (int)BN, CDIM, CDIM, 0);

  // ConvFFN
  k_layernorm<<<BN, CDIM, 0, stream>>>(x_res, n2g, n2b, xn2_bf);
  k_gemm_wmma<<<dim3(MLPH / 64, (int)(BN / 64), 1), 256, 0, stream>>>(
      xn2_bf, 0, fc1_bf, 0, fc1_b, nullptr, 0, 0, 0, xc_bf, 0, CH,
      (int)BN, MLPH, CDIM, 1 /*gelu*/);
  k_xtd_gather<<<cdiv(BN * DTD, 256), 256, 0, stream>>>(tdf_f, tkid, xc_bf);
  k_dwconv<<<cdiv(BN * CH, 256), 256, 0, stream>>>(xc_bf, dw_w, dw_b, s_bf);

  // out = x_res + s @ fc2 + b
  k_copy_f32<<<cdiv(BN * CDIM, 256), 256, 0, stream>>>(x_res, outF, BN * CDIM);
  k_gemm_wmma<<<dim3(CDIM / 64, (int)(BN / 64), 1), 256, 0, stream>>>(
      s_bf, 0, fc2_bf, 0, fc2_b, outF, 0, CDIM, 1, nullptr, 0, 0,
      (int)BN, CDIM, CH, 0);
}
